// STFT_21380347199929
// MI455X (gfx1250) — matrix operand order repaired and checked by reading, not compiled, not run on hardware
//
#include <hip/hip_runtime.h>

typedef __attribute__((ext_vector_type(16))) _Float16 v16h;
typedef __attribute__((ext_vector_type(8)))  _Float16 v8h;
typedef __attribute__((ext_vector_type(4)))  _Float16 v4h;
typedef __attribute__((ext_vector_type(8)))  float    v8f;
typedef __attribute__((ext_vector_type(4)))  float    v4f;

#define NFFT      2048
#define HOP       512
#define PAD       1024
#define L_IN      160000
#define T_FRAMES  313
#define F_BINS    1025
#define N_TOT     2050                 // real (1025) + imag (1025) stacked as GEMM-N
#define BC        64                   // 32 batch * 2 channels
#define MT        32                   // frames per M-tile
#define TILES_PER_ROW 10               // ceil(313/32)
#define SMP       ((MT - 1) * HOP + NFFT)          // 17920 samples staged (35.8 KB f16)
#define OUT_HALF  (BC * T_FRAMES * F_BINS)         // 20,532,800 floats (real block)
#define W_HALFS   ((size_t)N_TOT * NFFT)           // 4,198,400 f16 = 8.4 MB

// ---------------------------------------------------------------------------
// Prep: W_real/W_imag (fp32) -> stacked f16 B-matrix in workspace, row = GEMM
// column f (0..1024 real, 1025..2049 imag), contiguous K.
// ---------------------------------------------------------------------------
__global__ __launch_bounds__(256) void wcvt_kernel(
    const float* __restrict__ Wr, const float* __restrict__ Wi,
    _Float16* __restrict__ Wh)
{
    size_t q = (size_t)blockIdx.x * 256 + threadIdx.x;   // quad index
    size_t idx = q * 4;
    if (idx >= W_HALFS) return;
    const size_t split = (size_t)F_BINS * NFFT;          // 2,099,200 (mult of 4)
    const float* src = (idx < split) ? (Wr + idx) : (Wi + (idx - split));
    v4f f = *(const v4f*)src;
    v4h h;
    h[0] = (_Float16)f[0]; h[1] = (_Float16)f[1];
    h[2] = (_Float16)f[2]; h[3] = (_Float16)f[3];
    *(v4h*)(Wh + idx) = h;
}

// ---------------------------------------------------------------------------
// Main GEMM: per-wave 32(M) x 32(N) tile, 2x2 fragments, K=2048 step 32.
// A staged in LDS (f16, reflect-padded, converted once); B = f16 W from L2.
// ---------------------------------------------------------------------------
__global__ __launch_bounds__(256) void stft_wmma_f16w_kernel(
    const float* __restrict__ x,
    const _Float16* __restrict__ Wh,
    float* __restrict__ out)
{
    __shared__ _Float16 sm[SMP];

    const int bc   = blockIdx.x / TILES_PER_ROW;
    const int tile = blockIdx.x % TILES_PER_ROW;
    const int t0   = tile * MT;
    const float* __restrict__ xrow = x + (size_t)bc * L_IN;

    // ---- Stage A-tile samples into LDS as f16, applying reflect padding ----
    const int base = t0 * HOP - PAD;
    for (int i = threadIdx.x; i < SMP; i += 256) {   // SMP == 256*70, no divergence
        int s = base + i;
        s = (s < 0) ? -s : s;
        s = (s >= L_IN) ? (2 * L_IN - 2 - s) : s;
        sm[i] = (_Float16)xrow[s];
    }
    __syncthreads();

    const int lane = threadIdx.x & 31;
    const int wave = threadIdx.x >> 5;

    // Two N-fragments per wave: columns wave*32 + {0..15, 16..31}
    const int f0 = blockIdx.y * 256 + wave * 32 + (lane & 15);
    const int f1 = f0 + 16;
    const int f0c = (f0 < N_TOT - 1) ? f0 : (N_TOT - 1);
    const int f1c = (f1 < N_TOT - 1) ? f1 : (N_TOT - 1);

    const int koffB = (lane >> 4) * 16;   // B: lanes16-31 hold K=16..31
    const int koffA = (lane >> 4) * 16;   // A: lanes16-31 hold K=16..31
    const int am0   = (lane & 15);

    const _Float16* __restrict__ wb0 = Wh + (size_t)f0c * NFFT + koffB;
    const _Float16* __restrict__ wb1 = Wh + (size_t)f1c * NFFT + koffB;

    v8f acc00 = {}, acc01 = {}, acc10 = {}, acc11 = {};

    #pragma unroll 2
    for (int k0 = 0; k0 < NFFT; k0 += 32) {
        __builtin_prefetch(wb0 + k0 + 512, 0, 1);    // global_prefetch down K stream
        __builtin_prefetch(wb1 + k0 + 512, 0, 1);

        // B fragments: 16 contiguous f16 per lane (2x global_load_b128 each)
        v16h b0 = *(const v16h*)(wb0 + k0);
        v16h b1 = *(const v16h*)(wb1 + k0);

        // A fragments from LDS (two 8-half chunks per lane, +8 halves apart)
        const int ia0 = am0 * HOP + k0 + koffA;
        v8h a0lo = *(const v8h*)(sm + ia0);
        v8h a0hi = *(const v8h*)(sm + ia0 + 8);
        v16h a0 = __builtin_shufflevector(a0lo, a0hi,
                    0,1,2,3,4,5,6,7,8,9,10,11,12,13,14,15);

        const int ia1 = (am0 + 16) * HOP + k0 + koffA;
        v8h a1lo = *(const v8h*)(sm + ia1);
        v8h a1hi = *(const v8h*)(sm + ia1 + 8);
        v16h a1 = __builtin_shufflevector(a1lo, a1hi,
                    0,1,2,3,4,5,6,7,8,9,10,11,12,13,14,15);

        acc00 = __builtin_amdgcn_wmma_f32_16x16x32_f16(
                    false, a0, false, b0, (short)0, acc00, false, false);
        acc01 = __builtin_amdgcn_wmma_f32_16x16x32_f16(
                    false, a0, false, b1, (short)0, acc01, false, false);
        acc10 = __builtin_amdgcn_wmma_f32_16x16x32_f16(
                    false, a1, false, b0, (short)0, acc10, false, false);
        acc11 = __builtin_amdgcn_wmma_f32_16x16x32_f16(
                    false, a1, false, b1, (short)0, acc11, false, false);
    }

    // ---- Store: VGPR v -> rows v / v+8 (by lane half), column = lane&15 ----
    const int rofs = (lane >> 4) * 8;
    const size_t row0 = (size_t)(bc * T_FRAMES) * F_BINS;

    #pragma unroll
    for (int nf = 0; nf < 2; ++nf) {
        const int fc = (nf == 0) ? f0 : f1;
        if (fc >= N_TOT) continue;
        float* __restrict__ obase =
            (fc < F_BINS) ? (out + row0 + fc)
                          : (out + OUT_HALF + row0 + (fc - F_BINS));
        const v8f accA = (nf == 0) ? acc00 : acc01;
        const v8f accB = (nf == 0) ? acc10 : acc11;
        #pragma unroll
        for (int v = 0; v < 8; ++v) {
            const int t_a = t0 + v + rofs;            // frag rows 0..15
            if (t_a < T_FRAMES) obase[(size_t)t_a * F_BINS] = accA[v];
            const int t_b = t0 + 16 + v + rofs;       // frag rows 16..31
            if (t_b < T_FRAMES) obase[(size_t)t_b * F_BINS] = accB[v];
        }
    }
}

// ---------------------------------------------------------------------------
// Fallback (ws too small): previous version converting W f32->f16 in-loop.
// ---------------------------------------------------------------------------
__global__ __launch_bounds__(256) void stft_wmma_fb_kernel(
    const float* __restrict__ x,
    const float* __restrict__ Wr,
    const float* __restrict__ Wi,
    float* __restrict__ out)
{
    __shared__ _Float16 sm[SMP];

    const int bc   = blockIdx.x / TILES_PER_ROW;
    const int tile = blockIdx.x % TILES_PER_ROW;
    const int t0   = tile * MT;
    const float* __restrict__ xrow = x + (size_t)bc * L_IN;

    const int base = t0 * HOP - PAD;
    for (int i = threadIdx.x; i < SMP; i += 256) {
        int s = base + i;
        s = (s < 0) ? -s : s;
        s = (s >= L_IN) ? (2 * L_IN - 2 - s) : s;
        sm[i] = (_Float16)xrow[s];
    }
    __syncthreads();

    const int lane = threadIdx.x & 31;
    const int wave = threadIdx.x >> 5;
    const int fcol = blockIdx.y * 128 + wave * 16 + (lane & 15);
    const int fcl  = (fcol < N_TOT - 1) ? fcol : (N_TOT - 1);
    const float* __restrict__ wrow =
        (fcl < F_BINS) ? (Wr + (size_t)fcl * NFFT)
                       : (Wi + (size_t)(fcl - F_BINS) * NFFT);

    const int koffB = (lane >> 4) * 16;
    const int koffA = (lane >> 4) * 16;
    const int am0   = (lane & 15);

    v8f acc0 = {}, acc1 = {};

    #pragma unroll 2
    for (int k0 = 0; k0 < NFFT; k0 += 32) {
        const v4f* __restrict__ bp = (const v4f*)(wrow + k0 + koffB);
        v4f q0 = bp[0], q1 = bp[1], q2 = bp[2], q3 = bp[3];
        v16h b;
        #pragma unroll
        for (int j = 0; j < 4; ++j) {
            b[j] = (_Float16)q0[j];      b[4 + j]  = (_Float16)q1[j];
            b[8 + j] = (_Float16)q2[j];  b[12 + j] = (_Float16)q3[j];
        }
        const int ia0 = am0 * HOP + k0 + koffA;
        v8h a0lo = *(const v8h*)(sm + ia0);
        v8h a0hi = *(const v8h*)(sm + ia0 + 8);
        v16h a0 = __builtin_shufflevector(a0lo, a0hi,
                    0,1,2,3,4,5,6,7,8,9,10,11,12,13,14,15);
        const int ia1 = (am0 + 16) * HOP + k0 + koffA;
        v8h a1lo = *(const v8h*)(sm + ia1);
        v8h a1hi = *(const v8h*)(sm + ia1 + 8);
        v16h a1 = __builtin_shufflevector(a1lo, a1hi,
                    0,1,2,3,4,5,6,7,8,9,10,11,12,13,14,15);

        acc0 = __builtin_amdgcn_wmma_f32_16x16x32_f16(
                   false, a0, false, b, (short)0, acc0, false, false);
        acc1 = __builtin_amdgcn_wmma_f32_16x16x32_f16(
                   false, a1, false, b, (short)0, acc1, false, false);
    }

    if (fcol < N_TOT) {
        float* __restrict__ obase =
            (fcol < F_BINS)
                ? (out + (size_t)(bc * T_FRAMES) * F_BINS + fcol)
                : (out + OUT_HALF + (size_t)(bc * T_FRAMES) * F_BINS + (fcol - F_BINS));
        const int rofs = (lane >> 4) * 8;
        #pragma unroll
        for (int v = 0; v < 8; ++v) {
            const int t_a = t0 + v + rofs;
            if (t_a < T_FRAMES) obase[(size_t)t_a * F_BINS] = acc0[v];
            const int t_b = t0 + 16 + v + rofs;
            if (t_b < T_FRAMES) obase[(size_t)t_b * F_BINS] = acc1[v];
        }
    }
}

extern "C" void kernel_launch(void* const* d_in, const int* in_sizes, int n_in,
                              void* d_out, int out_size, void* d_ws, size_t ws_size,
                              hipStream_t stream) {
    (void)in_sizes; (void)n_in; (void)out_size;
    const float* x  = (const float*)d_in[0];
    const float* Wr = (const float*)d_in[1];
    const float* Wi = (const float*)d_in[2];
    float* out      = (float*)d_out;

    const size_t w_bytes = W_HALFS * sizeof(_Float16);   // 8.4 MB
    if (d_ws != nullptr && ws_size >= w_bytes) {
        _Float16* Wh = (_Float16*)d_ws;
        // 1) convert W to f16 once per launch (stream-ordered before GEMM)
        const int quads = (int)(W_HALFS / 4);            // 1,049,600
        wcvt_kernel<<<(quads + 255) / 256, 256, 0, stream>>>(Wr, Wi, Wh);
        // 2) GEMM: grid = (64 bc * 10 M-tiles, ceil(2050/256)) = (640, 9)
        dim3 grid(BC * TILES_PER_ROW, (N_TOT + 255) / 256);
        stft_wmma_f16w_kernel<<<grid, dim3(256), 0, stream>>>(x, Wh, out);
    } else {
        dim3 grid(BC * TILES_PER_ROW, (N_TOT + 127) / 128);  // (640, 17)
        stft_wmma_fb_kernel<<<grid, dim3(256), 0, stream>>>(x, Wr, Wi, out);
    }
}
